// ImageSequenceEncoder_84043920048158
// MI455X (gfx1250) — compile-verified, matrix operand order checked by reading, not run on hardware
//
#include <hip/hip_runtime.h>

typedef __attribute__((ext_vector_type(16))) _Float16 v16h;
typedef __attribute__((ext_vector_type(8)))  _Float16 v8h;
typedef __attribute__((ext_vector_type(8)))  float    v8f;

#define Tt    15
#define Hh    8
#define Ww    10
#define NPIX  80          // 8*10 = exactly 5 M-tiles of 16
#define Cc    32
#define NWAVE 8

// ---------------- packed-weight layout in d_ws (f16 elements) ----------------
// Per-tap fragments: frag index = (tap*NKS + ks)*NTILES + ntile, 512 f16 each.
// B layout mirrors 16-bit A layout: lane -> N=lane&15,
// cin = ks*32 + (slot>>3)*16 + ((lane>>4)&1)*8 + (slot&7)
#define NKS_W2   1   // Cin 32
#define NT_W2    2
#define NKS_TW1  16  // Cin 483 -> 512
#define NT_TW1   4
#define NKS_TW2  2   // Cin 64
#define NT_TW2   2
#define NKS_DW1  1   // Cin 32
#define NT_DW1   1
#define NKS_DW2  1   // Cin 16 -> 32
#define NT_DW2   1

#define OFF_W2   0
#define OFF_TW1  (OFF_W2  + 9*NKS_W2 *NT_W2 *512)
#define OFF_TW2  (OFF_TW1 + 9*NKS_TW1*NT_TW1*512)
#define OFF_DW1  (OFF_TW2 + 9*NKS_TW2*NT_TW2*512)
#define OFF_DW2  (OFF_DW1 + 9*NKS_DW1*NT_DW1*512)

// ---------------- dynamic LDS partition (bytes, all 16B aligned) ----------------
#define XSM_OFF    0            // [15][80] f16            = 2400
#define ZROW_OFF   2400         // 512 f16 zeros           = 1024
#define COMB_OFF   3424         // [80][512] f16           = 81920
#define FEAT1_OFF  85344        // [15][80][32] f16        = 76800 (dead after conv2)
#define TBUF_OFF   85344        // alias: [80][64] f16     = 10240
#define T2_OFF     95584        // [80][32] f16            = 5120
#define DBUF_OFF   100704       // [80][32] f16            = 5120
#define SMEM_BYTES 162144       // 2 WGs fit in 320KB WGP

// ============================================================================
// Weight pack: src (Cout,Cin,3,3) f32 -> per-tap WMMA B fragments.
// ============================================================================
__global__ void pack_weights_kernel(const float* __restrict__ src,
                                    _Float16* __restrict__ dst,
                                    int Cin, int Cout, int nks, int ntiles,
                                    int nelem) {
    int e = blockIdx.x * blockDim.x + threadIdx.x;
    if (e >= nelem) return;
    int frag   = e >> 9;
    int within = e & 511;
    int lane   = within >> 4;
    int slot   = within & 15;
    int tap    = frag / (nks * ntiles);
    int rem    = frag - tap * (nks * ntiles);
    int ks     = rem / ntiles;
    int ntile  = rem - ks * ntiles;
    int n   = (ntile << 4) + (lane & 15);
    int cin = (ks << 5) + ((slot >> 3) << 4) + (((lane >> 4) & 1) << 3) + (slot & 7);
    float v = 0.f;
    if (n < Cout && cin < Cin)
        v = src[(n * Cin + cin) * 9 + tap];
    dst[e] = (_Float16)v;
}

// ============================================================================
// One 16x16 output tile of a 3x3 SAME conv expressed as 9 shifted GEMMs.
// act: [pix][actCS] f16 in LDS (channel-contiguous per pixel, padded+zeroed).
// A-fragment per (tap,ks): two contiguous ds_load_b128 per lane; OOB pixels
// redirect to a zeroed LDS row (address cndmask, no exec divergence).
// ============================================================================
template <int NKS, int NTILES>
static __device__ inline v8f conv_tile_acc(const _Float16* __restrict__ act, int actCS,
                                           const _Float16* __restrict__ wpack,
                                           float bv, int mtile, int lane, int ntile,
                                           const _Float16* __restrict__ zrow) {
    v8f acc;
#pragma unroll
    for (int r = 0; r < 8; ++r) acc[r] = bv;

    int m   = lane & 15;
    int pix = (mtile << 4) + m;
    int h   = pix / Ww;
    int w   = pix - h * Ww;
    int kq  = ((lane >> 4) & 1) << 3;                 // 0 or 8
    const _Float16* wfrag = wpack + (ntile << 9) + ((lane & 31) << 4);

#pragma unroll
    for (int tap = 0; tap < 9; ++tap) {
        const int dh = tap / 3 - 1;
        const int dw = tap % 3 - 1;
        int hh = h + dh, ww = w + dw;
        bool valid = (hh >= 0) & (hh < Hh) & (ww >= 0) & (ww < Ww);
        const _Float16* base = valid ? (act + (hh * Ww + ww) * actCS + kq) : (zrow + kq);
#pragma unroll
        for (int ks = 0; ks < NKS; ++ks) {
            v8h lo = *(const v8h*)(base + (ks << 5));
            v8h hi = *(const v8h*)(base + (ks << 5) + 16);
            v16h a = __builtin_shufflevector(lo, hi, 0, 1, 2, 3, 4, 5, 6, 7,
                                             8, 9, 10, 11, 12, 13, 14, 15);
            v16h b = *(const v16h*)(wfrag + (((tap * NKS + ks) * NTILES) << 9));
            acc = __builtin_amdgcn_wmma_f32_16x16x32_f16(false, a, false, b,
                                                         (short)0, acc, false, false);
        }
    }
    return acc;
}

// Full stage: relu + bias, dst stored [pix][dstCS] at channel offset n.
template <int NKS, int NTILES>
static __device__ inline void conv_gemm(const _Float16* __restrict__ act, int actCS,
                                        const _Float16* __restrict__ wpack,
                                        const float* __restrict__ bias, int Cout,
                                        _Float16* __restrict__ dst, int dstCS,
                                        int lane, int wave,
                                        const _Float16* __restrict__ zrow) {
    for (int job = wave; job < 5 * NTILES; job += NWAVE) {
        int mtile = job / NTILES;
        int ntile = job - mtile * NTILES;
        int n = (ntile << 4) + (lane & 15);
        float bv = (n < Cout) ? bias[n] : 0.f;
        v8f acc = conv_tile_acc<NKS, NTILES>(act, actCS, wpack, bv, mtile, lane, ntile, zrow);
        if (n < Cout) {
#pragma unroll
            for (int r = 0; r < 8; ++r) {
                int pix = (mtile << 4) + r + (((lane >> 4) & 1) << 3);
                float v = acc[r] > 0.f ? acc[r] : 0.f;
                dst[pix * dstCS + n] = (_Float16)v;
            }
        }
    }
}

// ============================================================================
// Fused per-batch pipeline: one workgroup (8 waves) per batch element.
// ============================================================================
__global__ void __launch_bounds__(256)
image_seq_encoder_kernel(const float* __restrict__ x,
                         const float* __restrict__ w1, const float* __restrict__ b1,
                         const float* __restrict__ b2,
                         const float* __restrict__ tb1, const float* __restrict__ tb2,
                         const float* __restrict__ db1, const float* __restrict__ db2,
                         const _Float16* __restrict__ wp,
                         float* __restrict__ out) {
    extern __shared__ char smem[];
    _Float16* xsm   = (_Float16*)(smem + XSM_OFF);   // [15][80]
    _Float16* zrow  = (_Float16*)(smem + ZROW_OFF);  // 512 zeros
    _Float16* comb  = (_Float16*)(smem + COMB_OFF);  // [80][512]
    _Float16* feat1 = (_Float16*)(smem + FEAT1_OFF); // [15][80][32]
    _Float16* tbuf  = (_Float16*)(smem + TBUF_OFF);  // [80][64]
    _Float16* t2buf = (_Float16*)(smem + T2_OFF);    // [80][32]
    _Float16* dbuf  = (_Float16*)(smem + DBUF_OFF);  // [80][32]

    int tid  = threadIdx.x;
    int lane = tid & 31;
    int wave = tid >> 5;
    int b    = blockIdx.x;

    // ---- phase 1: stage x, zero the zero-row and comb channel padding ----
    const float* xb = x + (size_t)b * (Tt * NPIX);
    for (int i = tid; i < Tt * NPIX; i += 256) xsm[i] = (_Float16)xb[i];
    for (int i = tid; i < 512; i += 256) zrow[i] = (_Float16)0.f;
    for (int i = tid; i < NPIX * 29; i += 256) {        // comb cols 483..511
        int p = i / 29, c = i - p * 29;
        comb[p * 512 + 483 + c] = (_Float16)0.f;
    }
    __syncthreads();

    // ---- phase 2: extra maps (comb cols 480..482) + conv1 (VALU) ----
    for (int p = tid; p < NPIX; p += 256) {
        float s = 0.f;
#pragma unroll
        for (int t = 0; t < Tt; ++t) s += (float)xsm[t * NPIX + p];
        float rec = (float)xsm[(Tt - 1) * NPIX + p];
        comb[p * 512 + 480] = (_Float16)(s * (1.f / (float)Tt));
        comb[p * 512 + 481] = (_Float16)rec;
        comb[p * 512 + 482] = (_Float16)(1.f - rec);
    }
    for (int idx = tid; idx < Tt * Cc * NPIX; idx += 256) {
        int t   = idx / (Cc * NPIX);
        int rem = idx - t * (Cc * NPIX);
        int c   = rem / NPIX;
        int p   = rem - c * NPIX;
        int h   = p / Ww;
        int w   = p - h * Ww;
        float acc = b1[c];
#pragma unroll
        for (int kh = 0; kh < 3; ++kh) {
#pragma unroll
            for (int kw = 0; kw < 3; ++kw) {
                int hh = h + kh - 1, ww2 = w + kw - 1;
                if (hh >= 0 && hh < Hh && ww2 >= 0 && ww2 < Ww)
                    acc += (float)xsm[t * NPIX + hh * Ww + ww2] * w1[c * 9 + kh * 3 + kw];
            }
        }
        feat1[(t * NPIX + p) * Cc + c] = (_Float16)fmaxf(acc, 0.f);
    }
    __syncthreads();

    // ---- conv2 per frame: dst channel block f*32 of comb (stride 512) ----
    for (int job = wave; job < 150; job += NWAVE) {
        int f     = job / 10;
        int sub   = job - f * 10;
        int mtile = sub >> 1;
        int ntile = sub & 1;
        int n = (ntile << 4) + (lane & 15);
        float bv = b2[n];
        v8f acc = conv_tile_acc<NKS_W2, NT_W2>(feat1 + f * NPIX * Cc, Cc,
                                               wp + OFF_W2, bv, mtile, lane, ntile, zrow);
#pragma unroll
        for (int r = 0; r < 8; ++r) {
            int pix = (mtile << 4) + r + (((lane >> 4) & 1) << 3);
            float v = acc[r] > 0.f ? acc[r] : 0.f;
            comb[pix * 512 + f * Cc + n] = (_Float16)v;
        }
    }
    __syncthreads();

    // ---- tconv1: K = 512 (483 real), 64 out. Dominant GEMM: 144 WMMA/tile ----
    conv_gemm<NKS_TW1, NT_TW1>(comb, 512, wp + OFF_TW1, tb1, 64, tbuf, 64, lane, wave, zrow);
    __syncthreads();

    // ---- tconv2: 64 -> 32 ----
    conv_gemm<NKS_TW2, NT_TW2>(tbuf, 64, wp + OFF_TW2, tb2, 32, t2buf, 32, lane, wave, zrow);
    __syncthreads();

    // ---- dconv1: 32 -> 16 (dbuf cols 16..31 zeroed in same phase) ----
    for (int i = tid; i < NPIX * 16; i += 256) {
        int p = i >> 4, c = i & 15;
        dbuf[p * 32 + 16 + c] = (_Float16)0.f;
    }
    conv_gemm<NKS_DW1, NT_DW1>(t2buf, 32, wp + OFF_DW1, db1, 16, dbuf, 32, lane, wave, zrow);
    __syncthreads();

    // ---- dconv2: 16 -> 1, sigmoid, write output ----
    for (int mtile = wave; mtile < 5; mtile += NWAVE) {
        int n = lane & 15;
        float bv = (n == 0) ? db2[0] : 0.f;
        v8f acc = conv_tile_acc<NKS_DW2, NT_DW2>(dbuf, 32, wp + OFF_DW2, bv,
                                                 mtile, lane, 0, zrow);
        if (n == 0) {
#pragma unroll
            for (int r = 0; r < 8; ++r) {
                int pix = (mtile << 4) + r + (((lane >> 4) & 1) << 3);
                out[(size_t)b * NPIX + pix] = 1.f / (1.f + __expf(-acc[r]));
            }
        }
    }
}

// ============================================================================
extern "C" void kernel_launch(void* const* d_in, const int* in_sizes, int n_in,
                              void* d_out, int out_size, void* d_ws, size_t ws_size,
                              hipStream_t stream) {
    const float* x   = (const float*)d_in[0];
    const float* w1  = (const float*)d_in[1];
    const float* b1  = (const float*)d_in[2];
    const float* w2  = (const float*)d_in[3];
    const float* b2  = (const float*)d_in[4];
    const float* tw1 = (const float*)d_in[5];
    const float* tb1 = (const float*)d_in[6];
    const float* tw2 = (const float*)d_in[7];
    const float* tb2 = (const float*)d_in[8];
    const float* dw1 = (const float*)d_in[9];
    const float* db1 = (const float*)d_in[10];
    const float* dw2 = (const float*)d_in[11];
    const float* db2 = (const float*)d_in[12];
    float* out = (float*)d_out;
    _Float16* wp = (_Float16*)d_ws;

    int B = in_sizes[0] / (Tt * Hh * Ww);

    // Repack weights into per-tap WMMA B-fragment order (~0.66 MB, L2-resident)
    {
        int n;
        n = 9 * NKS_W2 * NT_W2 * 512;
        pack_weights_kernel<<<(n + 255) / 256, 256, 0, stream>>>(w2,  wp + OFF_W2,  Cc,  Cc, NKS_W2,  NT_W2,  n);
        n = 9 * NKS_TW1 * NT_TW1 * 512;
        pack_weights_kernel<<<(n + 255) / 256, 256, 0, stream>>>(tw1, wp + OFF_TW1, 483, 64, NKS_TW1, NT_TW1, n);
        n = 9 * NKS_TW2 * NT_TW2 * 512;
        pack_weights_kernel<<<(n + 255) / 256, 256, 0, stream>>>(tw2, wp + OFF_TW2, 64,  32, NKS_TW2, NT_TW2, n);
        n = 9 * NKS_DW1 * NT_DW1 * 512;
        pack_weights_kernel<<<(n + 255) / 256, 256, 0, stream>>>(dw1, wp + OFF_DW1, 32,  16, NKS_DW1, NT_DW1, n);
        n = 9 * NKS_DW2 * NT_DW2 * 512;
        pack_weights_kernel<<<(n + 255) / 256, 256, 0, stream>>>(dw2, wp + OFF_DW2, 16,  1,  NKS_DW2, NT_DW2, n);
    }

    image_seq_encoder_kernel<<<B, 256, SMEM_BYTES, stream>>>(
        x, w1, b1, b2, tb1, tb2, db1, db2, wp, out);
}
